// ScaledDotProductAttention_57200374448238
// MI455X (gfx1250) — matrix operand order repaired and checked by reading, not compiled, not run on hardware
//
#include <hip/hip_runtime.h>

#define S_LEN 2048
#define Dh 64
#define BH 64
#define NW 4

typedef __attribute__((ext_vector_type(16))) __bf16 v16bf;
typedef __attribute__((ext_vector_type(8)))  __bf16 v8bf;
typedef __attribute__((ext_vector_type(4)))  __bf16 v4bf;
typedef __attribute__((ext_vector_type(8)))  float  v8f;

// ---------------------------------------------------------------------------
// Pre-pass: K -> bf16 (same layout), V -> bf16 transposed per 64-key tile:
//   Vt[bh][tile][dim(64)][key(64)]
// ---------------------------------------------------------------------------
__global__ __launch_bounds__(256)
void preconv_kernel(const float* __restrict__ K, const float* __restrict__ V,
                    __bf16* __restrict__ Kb, __bf16* __restrict__ Vt)
{
  const size_t idx = (size_t)blockIdx.x * 256 + threadIdx.x;   // float4 slot
  const float4 kv = *((const float4*)K + idx);
  v4bf k4 = { (__bf16)kv.x, (__bf16)kv.y, (__bf16)kv.z, (__bf16)kv.w };
  *((v4bf*)Kb + idx) = k4;

  const float4 vv = *((const float4*)V + idx);
  const size_t e   = idx * 4;
  const size_t bh  = e / (S_LEN * Dh);
  const size_t rem = e % (S_LEN * Dh);
  const int key = (int)(rem / Dh);
  const int d   = (int)(rem % Dh);
  __bf16* dst = Vt + bh * (size_t)(S_LEN * Dh)
                   + (size_t)(key >> 6) * (64 * 64)
                   + (size_t)d * 64 + (key & 63);
  dst[0]   = (__bf16)vv.x;
  dst[64]  = (__bf16)vv.y;
  dst[128] = (__bf16)vv.z;
  dst[192] = (__bf16)vv.w;
}

__device__ __forceinline__ v16bf ld_frag16(const __bf16* p) {
  v8bf lo = *(const v8bf*)(const void*)p;
  v8bf hh = *(const v8bf*)(const void*)(p + 16);
  return __builtin_shufflevector(lo, hh, 0,1,2,3,4,5,6,7,8,9,10,11,12,13,14,15);
}

// Async global->LDS 16B copy (GVS form: SGPR64 base + per-lane voffset).
// LDS operand: low 32 bits of the generic pointer.
__device__ __forceinline__ void async_ld_b128(unsigned lds_off,
                                              unsigned long long sbase,
                                              unsigned voff) {
  asm volatile("global_load_async_to_lds_b128 %0, %1, %2"
               :: "v"(lds_off), "v"(voff), "s"(sbase) : "memory");
}
__device__ __forceinline__ void wait_async0() {
  asm volatile("s_wait_asynccnt 0" ::: "memory");
}

template<bool PRE>
__global__ __launch_bounds__(128)
void fa_fwd_kernel(const float* __restrict__ Q, const float* __restrict__ K,
                   const float* __restrict__ V, const unsigned char* __restrict__ Msk,
                   const __bf16* __restrict__ Kb, const __bf16* __restrict__ Vt,
                   float* __restrict__ O)
{
  __shared__ __bf16 sK[2][64][72];         // [buf][key][dim]
  __shared__ __bf16 sV[2][64][72];         // [buf][dim][key]
  __shared__ __bf16 sP[NW][16][72];        // per-wave P staging [row][key]
  __shared__ unsigned char sMt[64][72];    // mask tile TRANSPOSED: [key][q_local]

  const int t    = threadIdx.x;
  const int lane = t & 31;
  const int wv   = t >> 5;
  const int hi   = lane >> 4;
  const int m    = lane & 15;
  const int bh   = blockIdx.y;
  const int qblk = blockIdx.x * 64;
  const int q0   = qblk + wv * 16;

  const size_t qkv_bh = (size_t)bh * S_LEN * Dh;
  const size_t msk_bh = (size_t)bh * S_LEN * S_LEN;

  // staging geometry for this thread (4 x 16B chunks per matrix per tile)
  int srow[4], sc8[4];
  #pragma unroll
  for (int i = 0; i < 4; ++i) {
    const int c = i * 128 + t;
    srow[i] = c >> 3;
    sc8[i]  = (c & 7) << 3;
  }

  // ---- Q A-fragments, pre-scaled by 1/sqrt(64) ----
  const float* qrow = Q + qkv_bh + (size_t)(q0 + m) * Dh;
  v16bf aq[2];
  #pragma unroll
  for (int kc = 0; kc < 2; ++kc) {
    v16bf tv;
    const int base = kc * 32 + hi * 8;
    #pragma unroll
    for (int c = 0; c < 2; ++c) {
      const float4 f0 = *(const float4*)(qrow + base + c * 16);
      const float4 f1 = *(const float4*)(qrow + base + c * 16 + 4);
      const int e = c * 8;
      tv[e+0] = (__bf16)(f0.x * 0.125f); tv[e+1] = (__bf16)(f0.y * 0.125f);
      tv[e+2] = (__bf16)(f0.z * 0.125f); tv[e+3] = (__bf16)(f0.w * 0.125f);
      tv[e+4] = (__bf16)(f1.x * 0.125f); tv[e+5] = (__bf16)(f1.y * 0.125f);
      tv[e+6] = (__bf16)(f1.z * 0.125f); tv[e+7] = (__bf16)(f1.w * 0.125f);
    }
    aq[kc] = tv;
  }

  float mi[8], li[8];
  v8f oacc[4];
  #pragma unroll
  for (int r = 0; r < 8; ++r) { mi[r] = -1e30f; li[r] = 0.0f; }
  #pragma unroll
  for (int nt = 0; nt < 4; ++nt) oacc[nt] = (v8f){0,0,0,0,0,0,0,0};

  // ---- prologue: async-copy tile 0 into buffer 0 (PRE path) ----
  if constexpr (PRE) {
    const unsigned long long kb0 = (unsigned long long)(uintptr_t)(Kb + qkv_bh);
    const unsigned long long vb0 = (unsigned long long)(uintptr_t)(Vt + qkv_bh);
    #pragma unroll
    for (int i = 0; i < 4; ++i) {
      const unsigned goff = (unsigned)((srow[i] * 64 + sc8[i]) * 2);
      async_ld_b128((unsigned)(uintptr_t)(void*)&sK[0][srow[i]][sc8[i]], kb0, goff);
      async_ld_b128((unsigned)(uintptr_t)(void*)&sV[0][srow[i]][sc8[i]], vb0, goff);
    }
  }

  for (int j0 = 0; j0 < S_LEN; j0 += 64) {
    const int b = (j0 >> 6) & 1;

    // ---- mask tile j -> registers (global, coalesced) ----
    unsigned int w[2][4];
    int row4[2], col4[2];
    {
      const unsigned char* mb = Msk + msk_bh + (size_t)qblk * S_LEN + j0;
      #pragma unroll
      for (int i = 0; i < 2; ++i) {
        const int bi = i * 128 + t;          // 256 blocks of 4 rows x 4 cols
        row4[i] = (bi >> 4) << 2;
        col4[i] = (bi & 15) << 2;
        #pragma unroll
        for (int j = 0; j < 4; ++j)
          w[i][j] = *(const unsigned int*)(const void*)
                        (mb + (size_t)(row4[i] + j) * S_LEN + col4[i]);
      }
    }

    if constexpr (PRE) wait_async0();   // this wave's tile-j K/V copies landed
    __syncthreads();                    // barrier A: all waves' copies done,
                                        // previous compute done with sMt & buf b^1

    // ---- store transposed mask tile ----
    #pragma unroll
    for (int i = 0; i < 2; ++i) {
      #pragma unroll
      for (int c = 0; c < 4; ++c) {     // 4x4 byte transpose
        const unsigned int out =
            ((w[i][0] >> (8 * c)) & 0xffu)         |
            (((w[i][1] >> (8 * c)) & 0xffu) << 8)  |
            (((w[i][2] >> (8 * c)) & 0xffu) << 16) |
            (((w[i][3] >> (8 * c)) & 0xffu) << 24);
        *(unsigned int*)(void*)&sMt[col4[i] + c][row4[i]] = out;
      }
    }

    // ---- kick off next tile / or stage synchronously (fallback) ----
    if constexpr (PRE) {
      if (j0 + 64 < S_LEN) {
        const int nb = b ^ 1;
        const unsigned long long kbn =
            (unsigned long long)(uintptr_t)(Kb + qkv_bh + (size_t)(j0 + 64) * Dh);
        const unsigned long long vbn =
            (unsigned long long)(uintptr_t)(Vt + qkv_bh + (size_t)((j0 >> 6) + 1) * (64 * 64));
        #pragma unroll
        for (int i = 0; i < 4; ++i) {
          const unsigned goff = (unsigned)((srow[i] * 64 + sc8[i]) * 2);
          async_ld_b128((unsigned)(uintptr_t)(void*)&sK[nb][srow[i]][sc8[i]], kbn, goff);
          async_ld_b128((unsigned)(uintptr_t)(void*)&sV[nb][srow[i]][sc8[i]], vbn, goff);
        }
      }
    } else {
      const float* kbase = K + qkv_bh + (size_t)j0 * Dh;
      const float* vbase = V + qkv_bh + (size_t)j0 * Dh;
      float4 tk[4];
      #pragma unroll
      for (int g = 0; g < 2; ++g) {
        #pragma unroll
        for (int i = 0; i < 4; ++i) {
          const int idx = (g * 4 + i) * 128 + t;
          tk[i] = *(const float4*)(kbase + (idx >> 4) * Dh + ((idx & 15) << 2));
        }
        #pragma unroll
        for (int i = 0; i < 4; ++i) {
          const int idx = (g * 4 + i) * 128 + t;
          v4bf k4 = { (__bf16)tk[i].x, (__bf16)tk[i].y, (__bf16)tk[i].z, (__bf16)tk[i].w };
          *(v4bf*)(void*)&sK[b][idx >> 4][(idx & 15) << 2] = k4;
        }
        #pragma unroll
        for (int i = 0; i < 4; ++i) {
          const int idx = (g * 4 + i) * 128 + t;
          tk[i] = *(const float4*)(vbase + (idx >> 4) * Dh + ((idx & 15) << 2));
        }
        #pragma unroll
        for (int i = 0; i < 4; ++i) {
          const int idx = (g * 4 + i) * 128 + t;
          const int d4 = (idx & 15) << 2, key = idx >> 4;
          sV[b][d4 + 0][key] = (__bf16)tk[i].x;
          sV[b][d4 + 1][key] = (__bf16)tk[i].y;
          sV[b][d4 + 2][key] = (__bf16)tk[i].z;
          sV[b][d4 + 3][key] = (__bf16)tk[i].w;
        }
      }
    }
    __syncthreads();                    // barrier B: sMt (+fallback K/V) visible

    // ---- prefetch next mask tile ----
    if (j0 + 64 < S_LEN) {
      const unsigned char* mbn = Msk + msk_bh + (size_t)qblk * S_LEN + (j0 + 64);
      __builtin_prefetch(mbn + (size_t)(t & 63) * S_LEN, 0, 1);
      if constexpr (!PRE) {
        const float* kbn = K + qkv_bh + (size_t)(j0 + 64) * Dh;
        const float* vbn = V + qkv_bh + (size_t)(j0 + 64) * Dh;
        __builtin_prefetch(kbn + t * 32, 0, 1);
        __builtin_prefetch(vbn + t * 32, 0, 1);
      }
    }

    // ---- per-lane mask words: one b64 per 16-col tile ----
    unsigned long long mm[4];
    #pragma unroll
    for (int nt = 0; nt < 4; ++nt)
      mm[nt] = *(const unsigned long long*)(const void*)
                   &sMt[nt * 16 + m][wv * 16 + hi * 8];

    // ---- scores + mask bias ----
    float sc[4][8];
    #pragma unroll
    for (int nt = 0; nt < 4; ++nt) {
      v8f acc = (v8f){0,0,0,0,0,0,0,0};
      #pragma unroll
      for (int kc = 0; kc < 2; ++kc) {
        v16bf bk = ld_frag16(&sK[b][nt * 16 + m][kc * 32 + hi * 8]);
        acc = __builtin_amdgcn_wmma_f32_16x16x32_bf16(
                  false, aq[kc], false, bk, (short)0, acc, false, false);
      }
      const unsigned int mlo = (unsigned int)mm[nt];
      const unsigned int mhi = (unsigned int)(mm[nt] >> 32);
      #pragma unroll
      for (int r = 0; r < 8; ++r) {
        const unsigned int word = (r < 4) ? mlo : mhi;
        const float f = (float)((word >> (8 * (r & 3))) & 0xffu);  // v_cvt_f32_ubyteN
        sc[nt][r] = acc[r] + __builtin_fmaf(f, 1e30f, -1e30f);     // 0 or -1e30 bias
      }
    }

    // ---- online softmax ----
    float corr[8], rsum[8];
    #pragma unroll
    for (int r = 0; r < 8; ++r) {
      float rm = fmaxf(fmaxf(sc[0][r], sc[1][r]), fmaxf(sc[2][r], sc[3][r]));
      #pragma unroll
      for (int off = 1; off < 16; off <<= 1)
        rm = fmaxf(rm, __shfl_xor(rm, off, 32));
      const float mn = fmaxf(mi[r], rm);
      corr[r] = __expf(mi[r] - mn);
      mi[r] = mn;
      float ls = 0.0f;
      #pragma unroll
      for (int nt = 0; nt < 4; ++nt) {
        const float p = __expf(sc[nt][r] - mn);
        ls += p;
        sP[wv][r + hi * 8][nt * 16 + m] = (__bf16)p;
      }
      rsum[r] = ls;
    }
    #pragma unroll
    for (int r = 0; r < 8; ++r) {
      float ls = rsum[r];
      #pragma unroll
      for (int off = 1; off < 16; off <<= 1)
        ls += __shfl_xor(ls, off, 32);
      li[r] = li[r] * corr[r] + ls;
    }
    #pragma unroll
    for (int nt = 0; nt < 4; ++nt)
      #pragma unroll
      for (int r = 0; r < 8; ++r)
        oacc[nt][r] *= corr[r];

    // ---- P A-fragments ----
    v16bf ap[2];
    #pragma unroll
    for (int kc = 0; kc < 2; ++kc) {
      const __bf16* pp = &sP[wv][m][kc * 32 + hi * 8];
      v8bf lo = *(const v8bf*)(const void*)pp;
      v8bf hh = *(const v8bf*)(const void*)(pp + 16);
      ap[kc] = __builtin_shufflevector(lo, hh, 0,1,2,3,4,5,6,7,8,9,10,11,12,13,14,15);
    }

    // ---- O += P @ V ----
    #pragma unroll
    for (int nt = 0; nt < 4; ++nt) {
      #pragma unroll
      for (int kc = 0; kc < 2; ++kc) {
        v16bf bv = ld_frag16(&sV[b][nt * 16 + m][kc * 32 + hi * 8]);
        oacc[nt] = __builtin_amdgcn_wmma_f32_16x16x32_bf16(
                       false, ap[kc], false, bv, (short)0, oacc[nt], false, false);
      }
    }
  }

  // ---- epilogue ----
  float* orow = O + qkv_bh + (size_t)(q0 + hi * 8) * Dh;
  #pragma unroll
  for (int r = 0; r < 8; ++r) {
    const float rl = 1.0f / li[r];
    #pragma unroll
    for (int nt = 0; nt < 4; ++nt)
      orow[(size_t)r * Dh + nt * 16 + m] = oacc[nt][r] * rl;
  }
}

extern "C" void kernel_launch(void* const* d_in, const int* in_sizes, int n_in,
                              void* d_out, int out_size, void* d_ws, size_t ws_size,
                              hipStream_t stream) {
  const float*         Q = (const float*)d_in[0];
  const float*         K = (const float*)d_in[1];
  const float*         V = (const float*)d_in[2];
  const unsigned char* M = (const unsigned char*)d_in[3];
  float*               O = (float*)d_out;

  dim3 grid(S_LEN / 64, BH);
  dim3 block(128);

  const size_t mat_elems = (size_t)BH * S_LEN * Dh;           // 8.4M
  const size_t need      = 2 * mat_elems * sizeof(__bf16);    // 33.5 MB

  if (ws_size >= need) {
    __bf16* Kb = (__bf16*)d_ws;
    __bf16* Vt = Kb + mat_elems;
    const int nblocks = (int)(mat_elems / 4 / 256);
    preconv_kernel<<<nblocks, 256, 0, stream>>>(K, V, Kb, Vt);
    fa_fwd_kernel<true><<<grid, block, 0, stream>>>(Q, K, V, M, Kb, Vt, O);
  } else {
    fa_fwd_kernel<false><<<grid, block, 0, stream>>>(Q, K, V, M, nullptr, nullptr, O);
  }
}
